// GModule_27711128993993
// MI455X (gfx1250) — matrix operand order repaired and checked by reading, not compiled, not run on hardware
//
#include <hip/hip_runtime.h>
#include <hip/hip_bf16.h>

typedef _Float16 f16;
typedef __attribute__((ext_vector_type(16))) _Float16 v16h;
typedef __attribute__((ext_vector_type(8)))  float    v8f;
typedef int vi4 __attribute__((vector_size(16)));   // matches builtin's V4i param

#define BM 128
#define BN 64
#define BK 32

#if defined(__has_builtin)
#  if __has_builtin(__builtin_amdgcn_global_load_async_to_lds_b128)
#    define USE_ASYNC_LDS 1
#  endif
#endif
#ifndef USE_ASYNC_LDS
#  define USE_ASYNC_LDS 0
#endif

__device__ __forceinline__ void copy16_to_lds(const f16* g, f16* l) {
#if USE_ASYNC_LDS
  __builtin_amdgcn_global_load_async_to_lds_b128(
      (__attribute__((address_space(1))) vi4*)g,
      (__attribute__((address_space(3))) vi4*)l, 0, 0);
#else
  *(uint4*)l = *(const uint4*)g;
#endif
}
__device__ __forceinline__ void async_wait_all() {
#if USE_ASYNC_LDS
#  if defined(__has_builtin) && __has_builtin(__builtin_amdgcn_s_wait_asynccnt)
  __builtin_amdgcn_s_wait_asynccnt(0);
#  else
  asm volatile("s_wait_asynccnt 0x0" ::: "memory");
#  endif
#endif
}

// ---------------- reductions (wave32) ----------------
__device__ __forceinline__ float wave_sum(float v) {
#pragma unroll
  for (int o = 16; o > 0; o >>= 1) v += __shfl_down(v, o, 32);
  return v;
}
__device__ __forceinline__ float wave_max(float v) {
#pragma unroll
  for (int o = 16; o > 0; o >>= 1) v = fmaxf(v, __shfl_down(v, o, 32));
  return v;
}
__device__ __forceinline__ float block_sum(float v, float* sb) {
  __syncthreads();
  const int lane = threadIdx.x & 31, w = threadIdx.x >> 5, nw = blockDim.x >> 5;
  v = wave_sum(v);
  if (lane == 0) sb[w] = v;
  __syncthreads();
  float r = (threadIdx.x < (unsigned)nw) ? sb[threadIdx.x] : 0.0f;
  if (w == 0) { r = wave_sum(r); if (lane == 0) sb[0] = r; }
  __syncthreads();
  return sb[0];
}
__device__ __forceinline__ float block_max(float v, float* sb) {
  __syncthreads();
  const int lane = threadIdx.x & 31, w = threadIdx.x >> 5, nw = blockDim.x >> 5;
  v = wave_max(v);
  if (lane == 0) sb[w] = v;
  __syncthreads();
  float r = (threadIdx.x < (unsigned)nw) ? sb[threadIdx.x] : -1e30f;
  if (w == 0) { r = wave_max(r); if (lane == 0) sb[0] = r; }
  __syncthreads();
  return sb[0];
}

// ---------------- WMMA GEMM (A f16 [M,K], B f16 [N,K]; C = alpha*A*B^T) ----------------
// Optional: +bias[n], +residual (f32), relu, and f16 copy of C (row-major or transposed).
// Requires M%128==0, N%64==0, K%32==0 (true for all shapes used here).
__global__ __launch_bounds__(256)
void gemm_nt_wmma(const f16* __restrict__ A, const f16* __restrict__ B,
                  const float* __restrict__ bias, const float* __restrict__ residual,
                  float* __restrict__ C, f16* __restrict__ Ch,
                  int M, int N, int K, float alpha, int relu, int transCh) {
  __shared__ f16 ldsA[2][BM * BK];
  __shared__ f16 ldsB[2][BN * BK];

  const int tid  = threadIdx.x;
  const int gm0  = blockIdx.y * BM;
  const int gn0  = blockIdx.x * BN;
  const int lane = tid & 31;
  const int wid  = tid >> 5;
  const int wm   = (wid & 3) * 32;   // 4 waves along M
  const int wn   = (wid >> 2) * 32;  // 2 waves along N
  const int rl   = lane & 15;
  const int hs   = lane >> 4;

  v8f acc[2][2];
#pragma unroll
  for (int i = 0; i < 2; ++i)
#pragma unroll
    for (int j = 0; j < 2; ++j)
#pragma unroll
      for (int e = 0; e < 8; ++e) acc[i][j][e] = 0.0f;

  // staging: 16-byte chunks, K-contiguous rows for both A and B
  const int srow = tid >> 2;          // 0..63
  const int sseg = (tid & 3) * 8;     // half offset within 32-half row

  auto stage = [&](int buf, int k0) {
    copy16_to_lds(A + (size_t)(gm0 + srow) * K + k0 + sseg,
                  &ldsA[buf][srow * BK + sseg]);
    copy16_to_lds(A + (size_t)(gm0 + srow + 64) * K + k0 + sseg,
                  &ldsA[buf][(srow + 64) * BK + sseg]);
    copy16_to_lds(B + (size_t)(gn0 + srow) * K + k0 + sseg,
                  &ldsB[buf][srow * BK + sseg]);
  };

  union V { uint4 u[2]; v16h h; };

  stage(0, 0);
  async_wait_all();
  __syncthreads();

  int buf = 0;
  for (int k0 = 0; k0 < K; k0 += BK) {
    const int nxt = buf ^ 1;
    if (k0 + BK < K) stage(nxt, k0 + BK);           // async DMA overlaps WMMA below
    if (k0 + 2 * BK < K) {
      __builtin_prefetch(A + (size_t)(gm0 + srow) * K + k0 + 2 * BK, 0, 1);
      __builtin_prefetch(B + (size_t)(gn0 + srow) * K + k0 + 2 * BK, 0, 1);
    }

    // A fragment: halves 0..7 -> K = 8*hs + j ; halves 8..15 -> K = 16 + 8*hs + (j-8)
    V a0, a1, b0, b1;
    {
      const f16* base = &ldsA[buf][(wm + rl) * BK + hs * 8];
      a0.u[0] = *(const uint4*)(base);
      a0.u[1] = *(const uint4*)(base + 16);
    }
    {
      const f16* base = &ldsA[buf][(wm + 16 + rl) * BK + hs * 8];
      a1.u[0] = *(const uint4*)(base);
      a1.u[1] = *(const uint4*)(base + 16);
    }
    // B fragment: halves 0..7 -> K = 8*hs + j ; halves 8..15 -> K = 16 + 8*hs + (j-8)
    {
      const f16* base = &ldsB[buf][(wn + rl) * BK + hs * 8];
      b0.u[0] = *(const uint4*)(base);
      b0.u[1] = *(const uint4*)(base + 16);
    }
    {
      const f16* base = &ldsB[buf][(wn + 16 + rl) * BK + hs * 8];
      b1.u[0] = *(const uint4*)(base);
      b1.u[1] = *(const uint4*)(base + 16);
    }

    acc[0][0] = __builtin_amdgcn_wmma_f32_16x16x32_f16(false, a0.h, false, b0.h, (short)0, acc[0][0], false, false);
    acc[0][1] = __builtin_amdgcn_wmma_f32_16x16x32_f16(false, a0.h, false, b1.h, (short)0, acc[0][1], false, false);
    acc[1][0] = __builtin_amdgcn_wmma_f32_16x16x32_f16(false, a1.h, false, b0.h, (short)0, acc[1][0], false, false);
    acc[1][1] = __builtin_amdgcn_wmma_f32_16x16x32_f16(false, a1.h, false, b1.h, (short)0, acc[1][1], false, false);

    async_wait_all();
    __syncthreads();
    buf = nxt;
  }

  // epilogue: C layout — vgpr e: M = e + 8*hs ; N = rl
#pragma unroll
  for (int i = 0; i < 2; ++i)
#pragma unroll
    for (int j = 0; j < 2; ++j) {
      const int n = gn0 + wn + j * 16 + rl;
      const float bv = bias ? bias[n] : 0.0f;
#pragma unroll
      for (int e = 0; e < 8; ++e) {
        const int m = gm0 + wm + i * 16 + hs * 8 + e;
        const size_t idx = (size_t)m * N + n;
        float v = acc[i][j][e] * alpha + bv;
        if (residual) v += residual[idx];
        if (relu) v = fmaxf(v, 0.0f);
        if (C) C[idx] = v;
        if (Ch) {
          if (!transCh) Ch[idx] = (f16)v;
          else          Ch[(size_t)n * M + m] = (f16)v;
        }
      }
    }
}

// ---------------- conversion kernels ----------------
__global__ __launch_bounds__(256) void cvt_f16(const float* __restrict__ x,
                                               f16* __restrict__ y, int n) {
  for (int i = blockIdx.x * blockDim.x + threadIdx.x; i < n; i += gridDim.x * blockDim.x)
    y[i] = (f16)x[i];
}
// y[c*R + r] = x[r*C + c]  (f16 transposed copy)
__global__ __launch_bounds__(256) void cvt_f16_T(const float* __restrict__ x,
                                                 f16* __restrict__ y, int R, int Cc) {
  const int n = R * Cc;
  for (int i = blockIdx.x * blockDim.x + threadIdx.x; i < n; i += gridDim.x * blockDim.x) {
    const int r = i / Cc, c = i - r * Cc;
    y[(size_t)c * R + r] = (f16)x[i];
  }
}

// ---------------- elementwise / reduction kernels ----------------
__global__ __launch_bounds__(256) void ln_rows(const float* __restrict__ x,
                                               float* __restrict__ y,
                                               f16* __restrict__ yh, int relu) {
  __shared__ float sb[8];
  const size_t base = (size_t)blockIdx.x * 256;
  float v = x[base + threadIdx.x];
  const float mean = block_sum(v, sb) * (1.0f / 256.0f);
  const float d = v - mean;
  const float var = block_sum(d * d, sb) * (1.0f / 256.0f);
  float o = d * rsqrtf(var + 1e-5f);
  if (relu) o = fmaxf(o, 0.0f);
  if (y)  y[base + threadIdx.x] = o;
  if (yh) yh[base + threadIdx.x] = (f16)o;
}

__global__ __launch_bounds__(256) void softmax_to_h(const float* __restrict__ s,
                                                    f16* __restrict__ ph, int cols) {
  __shared__ float sb[8];
  const size_t base = (size_t)blockIdx.x * cols;
  float m = -1e30f;
  for (int j = threadIdx.x; j < cols; j += blockDim.x) m = fmaxf(m, s[base + j]);
  m = block_max(m, sb);
  float sum = 0.0f;
  for (int j = threadIdx.x; j < cols; j += blockDim.x) sum += __expf(s[base + j] - m);
  sum = block_sum(sum, sb);
  const float inv = 1.0f / sum;
  for (int j = threadIdx.x; j < cols; j += blockDim.x)
    ph[base + j] = (f16)(__expf(s[base + j] - m) * inv);
}

__global__ __launch_bounds__(256) void row_lse_sub(float* __restrict__ la, int cols) {
  __shared__ float sb[8];
  const size_t base = (size_t)blockIdx.x * cols;
  float m = -1e30f;
  for (int j = threadIdx.x; j < cols; j += blockDim.x) m = fmaxf(m, la[base + j]);
  m = block_max(m, sb);
  float sum = 0.0f;
  for (int j = threadIdx.x; j < cols; j += blockDim.x) sum += __expf(la[base + j] - m);
  sum = block_sum(sum, sb);
  const float lse = m + logf(sum);
  for (int j = threadIdx.x; j < cols; j += blockDim.x) la[base + j] -= lse;
}

__global__ __launch_bounds__(256) void col_lse_sub(float* __restrict__ la, int rows, int cols) {
  __shared__ float sb[8];
  const int c = blockIdx.x;
  float m = -1e30f;
  for (int r = threadIdx.x; r < rows; r += blockDim.x) m = fmaxf(m, la[(size_t)r * cols + c]);
  m = block_max(m, sb);
  float sum = 0.0f;
  for (int r = threadIdx.x; r < rows; r += blockDim.x) sum += __expf(la[(size_t)r * cols + c] - m);
  sum = block_sum(sum, sb);
  const float lse = m + logf(sum);
  for (int r = threadIdx.x; r < rows; r += blockDim.x) la[(size_t)r * cols + c] -= lse;
}

__global__ __launch_bounds__(64) void zero_k(float* p, int n) {
  const int i = blockIdx.x * blockDim.x + threadIdx.x;
  if (i < n) p[i] = 0.0f;
}

__global__ __launch_bounds__(256) void reduce_meanstd(const float* __restrict__ x, int n,
                                                      float* __restrict__ sc) {
  __shared__ float sb[8];
  float s = 0.0f, s2 = 0.0f;
  for (int i = blockIdx.x * blockDim.x + threadIdx.x; i < n; i += gridDim.x * blockDim.x) {
    const float v = x[i]; s += v; s2 += v * v;
  }
  s  = block_sum(s, sb);
  s2 = block_sum(s2, sb);
  if (threadIdx.x == 0) { atomicAdd(&sc[0], s); atomicAdd(&sc[1], s2); }
}

__global__ __launch_bounds__(256) void normalize_inplace(float* __restrict__ x, int n,
                                                         const float* __restrict__ sc) {
  const float mean = sc[0] / (float)n;
  const float var  = sc[1] / (float)n - mean * mean;
  const float inv  = 1.0f / (sqrtf(fmaxf(var, 0.0f)) + 1e-5f);
  for (int i = blockIdx.x * blockDim.x + threadIdx.x; i < n; i += gridDim.x * blockDim.x)
    x[i] = (x[i] - mean) * inv;
}

__global__ __launch_bounds__(256) void match_loss_k(const float* __restrict__ la,
                                                    const int* __restrict__ ls,
                                                    const int* __restrict__ lt,
                                                    int nt, int n, float* __restrict__ sc) {
  __shared__ float sb[8];
  float acc = 0.0f;
  for (int i = blockIdx.x * blockDim.x + threadIdx.x; i < n; i += gridDim.x * blockDim.x) {
    const int r = i / nt, c = i - r * nt;
    const float p = __expf(la[i]);
    const float gt = (ls[r] == lt[c]) ? 1.0f : 0.0f;
    acc += fabsf(p - gt);
  }
  acc = block_sum(acc, sb);
  if (threadIdx.x == 0) atomicAdd(&sc[2], acc);
}

__global__ __launch_bounds__(256) void featnorm(const float* __restrict__ s2,
                                                const float* __restrict__ t2,
                                                f16* __restrict__ fh, int ns) {
  __shared__ float sb[8];
  const int row = blockIdx.x;
  const float* src = (row < ns) ? (s2 + (size_t)row * 256) : (t2 + (size_t)(row - ns) * 256);
  const float v = src[threadIdx.x];
  const float ss = block_sum(v * v, sb);
  fh[(size_t)row * 256 + threadIdx.x] = (f16)(v / (sqrtf(ss) + 1e-8f));
}

__global__ __launch_bounds__(256) void supcon_rows(const float* __restrict__ logits,
                                                   const int* __restrict__ ls,
                                                   const int* __restrict__ lt,
                                                   int n, int ns, float* __restrict__ sc) {
  __shared__ float sb[8];
  const int i = blockIdx.x;
  const int li = (i < ns) ? ls[i] : lt[i - ns];
  const float* row = logits + (size_t)i * n;
  float m = -1e30f;
  for (int j = threadIdx.x; j < n; j += blockDim.x) m = fmaxf(m, row[j]);
  m = block_max(m, sb);
  float se = 0.0f, ps = 0.0f, cnt = 0.0f;
  for (int j = threadIdx.x; j < n; j += blockDim.x) {
    if (j == i) continue;
    const float l = row[j];
    se += __expf(l - m);
    const int lj = (j < ns) ? ls[j] : lt[j - ns];
    if (lj == li) { ps += l; cnt += 1.0f; }
  }
  se  = block_sum(se, sb);
  ps  = block_sum(ps, sb);
  cnt = block_sum(cnt, sb);
  if (threadIdx.x == 0) {
    const float mlpp = ps / cnt - m - logf(se);
    atomicAdd(&sc[3], -mlpp);
  }
}

__global__ __launch_bounds__(32) void finalize_k(const float* __restrict__ sc,
                                                 float* __restrict__ out) {
  if (threadIdx.x == 0) out[0] = 1.0f * sc[2] + 0.1f * (sc[3] / 4096.0f);
}

// ---------------- host orchestration ----------------
extern "C" void kernel_launch(void* const* d_in, const int* in_sizes, int n_in,
                              void* d_out, int out_size, void* d_ws, size_t ws_size,
                              hipStream_t stream) {
  (void)in_sizes; (void)n_in; (void)out_size; (void)ws_size;
  const float* nodes_src = (const float*)d_in[0];
  const float* nodes_tgt = (const float*)d_in[1];
  const int*   lab_src   = (const int*)d_in[2];
  const int*   lab_tgt   = (const int*)d_in[3];
  const float* w1 = (const float*)d_in[4];  const float* b1 = (const float*)d_in[5];
  const float* w2 = (const float*)d_in[6];  const float* b2 = (const float*)d_in[7];
  const float* wq = (const float*)d_in[8];  const float* wk = (const float*)d_in[9];
  const float* wv = (const float*)d_in[10]; const float* wo = (const float*)d_in[11];
  const float* cq = (const float*)d_in[12]; const float* ck = (const float*)d_in[13];
  const float* cv = (const float*)d_in[14]; const float* co = (const float*)d_in[15];
  const float* Am = (const float*)d_in[16];

  const int NS = 2048, NT = 2048, D = 256, NF = 4096;
  const size_t ND = (size_t)NS * D;

  // ---- f32 workspace ----
  float* ws = (float*)d_ws;
  size_t off = 0;
  auto takeF = [&](size_t n) { float* p = ws + off; off += n; return p; };
  float* t1    = takeF(ND);
  float* t2    = takeF(ND);
  float* bsrc  = takeF(ND);
  float* btgt  = takeF(ND);
  float* bsrc2 = takeF(ND);
  float* btgt2 = takeF(ND);
  float* bS    = takeF((size_t)NS * NT);
  float* bM    = takeF((size_t)NS * NT);
  float* blog  = takeF((size_t)NF * NF);
  float* sc    = takeF(8);

  // ---- f16 workspace (after the f32 region) ----
  f16* wh = (f16*)(ws + off);
  size_t hoff = 0;
  auto takeH = [&](size_t n) { f16* p = wh + hoff; hoff += n; return p; };
  f16* xsrch  = takeH(ND);
  f16* xtgth  = takeH(ND);
  f16* w1t = takeH(65536); f16* w2t = takeH(65536);
  f16* wqt = takeH(65536); f16* wkt = takeH(65536);
  f16* wvt = takeH(65536); f16* wot = takeH(65536);
  f16* cqt = takeH(65536); f16* ckt = takeH(65536);
  f16* cvt_ = takeH(65536); f16* cot = takeH(65536);
  f16* amt = takeH(65536);
  f16* t2h    = takeH(ND);
  f16* bsrch  = takeH(ND);
  f16* btgth  = takeH(ND);
  f16* qh     = takeH(ND);
  f16* kh     = takeH(ND);
  f16* vth    = takeH(ND);         // transposed [D, Nk]
  f16* atth   = takeH(ND);
  f16* t1h    = takeH(ND);
  f16* bsrc2h = takeH(ND);
  f16* btgt2h = takeH(ND);
  f16* Ph     = takeH((size_t)NS * NT);
  f16* featsh = takeH((size_t)NF * D);

  auto gemm = [&](const f16* A_, const f16* B_, const float* bias, const float* res,
                  float* C_, f16* Ch_, int M, int N, int K,
                  float alpha, int relu, int transCh) {
    dim3 g(N / BN, M / BM);
    gemm_nt_wmma<<<g, 256, 0, stream>>>(A_, B_, bias, res, C_, Ch_, M, N, K,
                                        alpha, relu, transCh);
  };

  zero_k<<<1, 64, 0, stream>>>(sc, 8);

  // one-time f16 conversions (inputs + transposed weights)
  cvt_f16<<<512, 256, 0, stream>>>(nodes_src, xsrch, (int)ND);
  cvt_f16<<<512, 256, 0, stream>>>(nodes_tgt, xtgth, (int)ND);
  cvt_f16_T<<<64, 256, 0, stream>>>(w1, w1t, D, D);
  cvt_f16_T<<<64, 256, 0, stream>>>(w2, w2t, D, D);
  cvt_f16_T<<<64, 256, 0, stream>>>(wq, wqt, D, D);
  cvt_f16_T<<<64, 256, 0, stream>>>(wk, wkt, D, D);
  cvt_f16_T<<<64, 256, 0, stream>>>(wv, wvt, D, D);
  cvt_f16_T<<<64, 256, 0, stream>>>(wo, wot, D, D);
  cvt_f16_T<<<64, 256, 0, stream>>>(cq, cqt, D, D);
  cvt_f16_T<<<64, 256, 0, stream>>>(ck, ckt, D, D);
  cvt_f16_T<<<64, 256, 0, stream>>>(cv, cvt_, D, D);
  cvt_f16_T<<<64, 256, 0, stream>>>(co, cot, D, D);
  cvt_f16_T<<<64, 256, 0, stream>>>(Am, amt, D, D);

  // head_in: LN(ReLU(LN(x@w1+b1)) @ w2 + b2)
  auto head = [&](const f16* xh, float* outF, f16* outH) {
    gemm(xh, w1t, b1, nullptr, t1, nullptr, NS, D, D, 1.0f, 0, 0);
    ln_rows<<<NS, 256, 0, stream>>>(t1, t2, t2h, 1);
    gemm(t2h, w2t, b2, nullptr, t1, nullptr, NS, D, D, 1.0f, 0, 0);
    ln_rows<<<NS, 256, 0, stream>>>(t1, outF, outH, 0);
  };
  head(xsrch, bsrc, bsrch);
  head(xtgth, btgt, btgth);

  // attention: out = resid + softmax((x Wq)(kv Wk)^T / 16) (kv Wv) Wo
  auto attn = [&](const f16* xh, const f16* kvh,
                  const f16* Wqt, const f16* Wkt, const f16* Wvt, const f16* Wot,
                  const float* residF, float* outF, f16* outH, int Nq, int Nk) {
    gemm(xh,  Wqt, nullptr, nullptr, nullptr, qh,  Nq, D, D, 1.0f, 0, 0);
    gemm(kvh, Wkt, nullptr, nullptr, nullptr, kh,  Nk, D, D, 1.0f, 0, 0);
    gemm(kvh, Wvt, nullptr, nullptr, nullptr, vth, Nk, D, D, 1.0f, 0, 1); // f16 out transposed [D,Nk]
    gemm(qh, kh, nullptr, nullptr, bS, nullptr, Nq, Nk, D, 0.0625f, 0, 0); // 1/sqrt(256)
    softmax_to_h<<<Nq, 256, 0, stream>>>(bS, Ph, Nk);
    gemm(Ph, vth, nullptr, nullptr, nullptr, atth, Nq, D, Nk, 1.0f, 0, 0);
    gemm(atth, Wot, nullptr, residF, outF, outH, Nq, D, D, 1.0f, 0, 0);
  };

  // intra-domain (residual written back in place; f16 copy refreshed)
  attn(bsrch, bsrch, wqt, wkt, wvt, wot, bsrc, bsrc, bsrch, NS, NS);
  attn(btgth, btgth, wqt, wkt, wvt, wot, btgt, btgt, btgth, NT, NT);
  // cross-domain
  attn(bsrch, btgth, cqt, ckt, cvt_, cot, bsrc, bsrc2, bsrc2h, NS, NT);
  attn(btgth, bsrch, cqt, ckt, cvt_, cot, btgt, btgt2, btgt2h, NT, NS);

  // M = src2 @ A @ tgt2^T, instance-normalized
  gemm(bsrc2h, amt, nullptr, nullptr, nullptr, t1h, NS, D, D, 1.0f, 0, 0);
  gemm(t1h, btgt2h, nullptr, nullptr, bM, nullptr, NS, NT, D, 1.0f, 0, 0);
  const int nM = NS * NT;
  reduce_meanstd<<<1024, 256, 0, stream>>>(bM, nM, sc);
  normalize_inplace<<<4096, 256, 0, stream>>>(bM, nM, sc);

  // log-space Sinkhorn: 10 x (row lse sub, col lse sub)
  for (int it = 0; it < 10; ++it) {
    row_lse_sub<<<NS, 256, 0, stream>>>(bM, NT);
    col_lse_sub<<<NT, 256, 0, stream>>>(bM, NS, NT);
  }
  match_loss_k<<<4096, 256, 0, stream>>>(bM, lab_src, lab_tgt, NT, nM, sc);

  // SupCon on L2-normalized concat(src2, tgt2)
  featnorm<<<NF, 256, 0, stream>>>(bsrc2, btgt2, featsh, NS);
  gemm(featsh, featsh, nullptr, nullptr, blog, nullptr, NF, NF, D, 1.0f / 0.07f, 0, 0);
  supcon_rows<<<NF, 256, 0, stream>>>(blog, lab_src, lab_tgt, NF, NS, sc);

  finalize_k<<<1, 32, 0, stream>>>(sc, (float*)d_out);
}
